// EnhancedTemporalAttention_50474455662835
// MI455X (gfx1250) — matrix operand order repaired and checked by reading, not compiled, not run on hardware
//
#include <hip/hip_runtime.h>

typedef __bf16 bf16;
typedef __bf16 v16bf __attribute__((ext_vector_type(16)));
typedef __bf16 v8bf  __attribute__((ext_vector_type(8)));
typedef float  v8f   __attribute__((ext_vector_type(8)));

union ABu { v16bf v; v8bf h[2]; };

static constexpr int Bb  = 4;
static constexpr int Cc  = 512;
static constexpr int Nn  = 2048;
static constexpr int Hh  = 8;
static constexpr int Dd  = 64;
static constexpr int Gg  = 32;
static constexpr int CPG = Cc / Gg;          // 16 channels per group
static constexpr float EPSc   = 1e-4f;
static constexpr float SCALEc = 0.125f;      // 64^-0.5

__device__ __forceinline__ bf16 f2bf(float f) {
    unsigned u = __float_as_uint(f);
    u += 0x7FFFu + ((u >> 16) & 1u);         // round-to-nearest-even
    unsigned short hs = (unsigned short)(u >> 16);
    return __builtin_bit_cast(bf16, hs);
}

__device__ __forceinline__ v8f wmma_bf16(v16bf a, v16bf b, v8f c) {
    return __builtin_amdgcn_wmma_f32_16x16x32_bf16(false, a, false, b,
                                                   (short)0, c, false, false);
}

// reductions across the 16 lanes of a half-wave (xor masks stay within halves)
__device__ __forceinline__ float hmax16(float v) {
    v = fmaxf(v, __shfl_xor(v, 1, 32));
    v = fmaxf(v, __shfl_xor(v, 2, 32));
    v = fmaxf(v, __shfl_xor(v, 4, 32));
    v = fmaxf(v, __shfl_xor(v, 8, 32));
    return v;
}
__device__ __forceinline__ float hsum16(float v) {
    v += __shfl_xor(v, 1, 32);
    v += __shfl_xor(v, 2, 32);
    v += __shfl_xor(v, 4, 32);
    v += __shfl_xor(v, 8, 32);
    return v;
}

// ---------------- kernel 0: fp32 -> bf16 weight conversion ----------------
__global__ void eta_cvt_kernel(const float* __restrict__ wa, bf16* __restrict__ oa, int na,
                               const float* __restrict__ wb, bf16* __restrict__ ob, int nb) {
    int i = blockIdx.x * 256 + threadIdx.x;
    if (i < na) oa[i] = f2bf(wa[i]);
    if (i < nb) ob[i] = f2bf(wb[i]);
}

// ---------------- kernel 1: GroupNorm -> xnt[b][n][C] (bf16) ----------------
__global__ void eta_gn_kernel(const float* __restrict__ x,
                              const float* __restrict__ gamma,
                              const float* __restrict__ beta,
                              bf16* __restrict__ xnt) {
    int b = blockIdx.x / Gg;
    int g = blockIdx.x % Gg;
    const float* xg = x + ((size_t)b * Cc + (size_t)g * CPG) * Nn;
    const int M = CPG * Nn;                       // 32768 elements per group
    float s = 0.f, ss = 0.f;
    for (int j = threadIdx.x; j < M; j += 256) {
        float v = xg[j];
        s += v; ss += v * v;
    }
    __shared__ float rs[256], rss[256];
    rs[threadIdx.x] = s; rss[threadIdx.x] = ss;
    __syncthreads();
    for (int o = 128; o > 0; o >>= 1) {
        if (threadIdx.x < (unsigned)o) {
            rs[threadIdx.x]  += rs[threadIdx.x + o];
            rss[threadIdx.x] += rss[threadIdx.x + o];
        }
        __syncthreads();
    }
    float mean = rs[0] / (float)M;
    float var  = rss[0] / (float)M - mean * mean;
    float rstd = rsqrtf(var + EPSc);
    for (int j = threadIdx.x; j < M; j += 256) {
        int cl = j / Nn, t = j % Nn;
        int c  = g * CPG + cl;
        float v = (xg[j] - mean) * rstd * gamma[c] + beta[c];
        xnt[((size_t)b * Nn + t) * Cc + c] = f2bf(v);
    }
}

// ---- kernel 2: QKV GEMM (1 M-tile x 4 N-tiles per wave) ----
// -> Q[bh][n][64], K[bh][n][64], Vt[bh][64][n]
__global__ void eta_qkv_kernel(const bf16* __restrict__ W,    // [1536][512]
                               const bf16* __restrict__ xnt,  // [b][n][C]
                               bf16* __restrict__ Qb, bf16* __restrict__ Kb,
                               bf16* __restrict__ Vt) {
    int lane = threadIdx.x;
    int half = lane >> 4, lc = lane & 15;
    int nb = blockIdx.x * 64;                    // 4 N tiles of 16
    int mb = blockIdx.y * 16;
    int b  = blockIdx.z;
    v8f acc[4] = {};
    const bf16* arow = W   + (size_t)(mb + lc) * Cc;
    const bf16* bcol = xnt + ((size_t)b * Nn + nb + lc) * Cc;
#pragma unroll 2
    for (int cb = 0; cb < Cc; cb += 32) {
        ABu a;
        a.h[0] = *(const v8bf*)(arow + cb + 16 * half);
        a.h[1] = *(const v8bf*)(arow + cb + 8 + 16 * half);
#pragma unroll
        for (int nt = 0; nt < 4; ++nt) {
            v16bf bm = *(const v16bf*)(bcol + (size_t)nt * 16 * Cc + cb + 16 * half);
            acc[nt] = wmma_bf16(a.v, bm, acc[nt]);
        }
    }
    int which = mb >> 9;                         // uniform per tile: 0=q 1=k 2=v
#pragma unroll
    for (int nt = 0; nt < 4; ++nt) {
        int t = nb + nt * 16 + lc;
#pragma unroll
        for (int g = 0; g < 8; ++g) {
            int o = mb + g + 8 * half;
            int r = o & 511;
            int h = r >> 6, d = r & 63;
            size_t bh = (size_t)b * Hh + h;
            bf16 v = f2bf(acc[nt][g]);
            if (which == 0)       Qb[(bh * Nn + t) * Dd + d] = v;
            else if (which == 1)  Kb[(bh * Nn + t) * Dd + d] = v;
            else                  Vt[(bh * Dd + d) * Nn + t] = v;
        }
    }
}

// ---- kernel 3: flash attention, 32 q-rows (2 M-tiles) per wave ----
__global__ void eta_attn_kernel(const bf16* __restrict__ Qb,
                                const bf16* __restrict__ Kb,
                                const bf16* __restrict__ Vt,
                                bf16* __restrict__ outt) {   // [b][n][C]
    __shared__ bf16 Plds[32 * 32];
    int lane = threadIdx.x;
    int half = lane >> 4, lc = lane & 15;
    int qbase = blockIdx.x * 32;
    int bh = blockIdx.y;
    int b = bh / Hh, h = bh % Hh;
    const bf16* Qp = Qb + (size_t)bh * Nn * Dd;
    const bf16* Kp = Kb + (size_t)bh * Nn * Dd;
    const bf16* Vp = Vt + (size_t)bh * Dd * Nn;

    // Q A-operands: [mt][k-block of 32 within D=64]
    ABu aq[2][2];
#pragma unroll
    for (int mt = 0; mt < 2; ++mt) {
        const bf16* qrow = Qp + (size_t)(qbase + mt * 16 + lc) * Dd;
        aq[mt][0].h[0] = *(const v8bf*)(qrow + 0  + 16 * half);
        aq[mt][0].h[1] = *(const v8bf*)(qrow + 8  + 16 * half);
        aq[mt][1].h[0] = *(const v8bf*)(qrow + 32 + 16 * half);
        aq[mt][1].h[1] = *(const v8bf*)(qrow + 40 + 16 * half);
    }
    v8f oacc[2][4] = {};
    float m[2][8], l[2][8];
#pragma unroll
    for (int mt = 0; mt < 2; ++mt)
#pragma unroll
        for (int g = 0; g < 8; ++g) { m[mt][g] = -1e30f; l[mt][g] = 0.f; }

    for (int kb = 0; kb < Nn; kb += 32) {
        // ---- S = Q K^T for a 32(q) x 32(k) strip; K B-operands shared by both M tiles
        v8f s[2][2] = {};
        {
            const bf16* kr0 = Kp + (size_t)(kb + lc) * Dd;       // cols 0..15
            const bf16* kr1 = Kp + (size_t)(kb + 16 + lc) * Dd;  // cols 16..31
            v16bf b00 = *(const v16bf*)(kr0 + 0  + 16 * half);
            v16bf b01 = *(const v16bf*)(kr0 + 32 + 16 * half);
            v16bf b10 = *(const v16bf*)(kr1 + 0  + 16 * half);
            v16bf b11 = *(const v16bf*)(kr1 + 32 + 16 * half);
#pragma unroll
            for (int mt = 0; mt < 2; ++mt) {
                s[mt][0] = wmma_bf16(aq[mt][0].v, b00, s[mt][0]);
                s[mt][0] = wmma_bf16(aq[mt][1].v, b01, s[mt][0]);
                s[mt][1] = wmma_bf16(aq[mt][0].v, b10, s[mt][1]);
                s[mt][1] = wmma_bf16(aq[mt][1].v, b11, s[mt][1]);
            }
        }
        // ---- online softmax (each row lives across 16 lanes of a half-wave) ----
#pragma unroll
        for (int mt = 0; mt < 2; ++mt) {
#pragma unroll
            for (int g = 0; g < 8; ++g) {
                float a0 = s[mt][0][g] * SCALEc, a1 = s[mt][1][g] * SCALEc;
                float mx = hmax16(fmaxf(a0, a1));
                float mn = fmaxf(m[mt][g], mx);
                float corr = __expf(m[mt][g] - mn);
                float e0 = __expf(a0 - mn), e1 = __expf(a1 - mn);
                float rs = hsum16(e0 + e1);
                l[mt][g] = l[mt][g] * corr + rs;
                m[mt][g] = mn;
                oacc[mt][0][g] *= corr; oacc[mt][1][g] *= corr;
                oacc[mt][2][g] *= corr; oacc[mt][3][g] *= corr;
                int row = mt * 16 + g + 8 * half;
                Plds[row * 32 + lc]      = f2bf(e0);
                Plds[row * 32 + 16 + lc] = f2bf(e1);
            }
        }
        __syncthreads();
        // ---- O += P V; V B-operands shared by both M tiles ----
        ABu ap[2];
#pragma unroll
        for (int mt = 0; mt < 2; ++mt) {
            ap[mt].h[0] = *(const v8bf*)(&Plds[(mt * 16 + lc) * 32 + 16 * half]);
            ap[mt].h[1] = *(const v8bf*)(&Plds[(mt * 16 + lc) * 32 + 8 + 16 * half]);
        }
#pragma unroll
        for (int dt = 0; dt < 4; ++dt) {
            const bf16* vrow = Vp + (size_t)(dt * 16 + lc) * Nn + kb + 16 * half;
            v16bf bv = *(const v16bf*)vrow;
            oacc[0][dt] = wmma_bf16(ap[0].v, bv, oacc[0][dt]);
            oacc[1][dt] = wmma_bf16(ap[1].v, bv, oacc[1][dt]);
        }
        __syncthreads();
    }
    // ---- normalize and store to outt[b][q][h*64 + d] ----
#pragma unroll
    for (int mt = 0; mt < 2; ++mt) {
#pragma unroll
        for (int g = 0; g < 8; ++g) {
            float inv = 1.0f / l[mt][g];
            int q = qbase + mt * 16 + g + 8 * half;
            bf16* orow = outt + ((size_t)b * Nn + q) * Cc + h * Dd;
            orow[0  + lc] = f2bf(oacc[mt][0][g] * inv);
            orow[16 + lc] = f2bf(oacc[mt][1][g] * inv);
            orow[32 + lc] = f2bf(oacc[mt][2][g] * inv);
            orow[48 + lc] = f2bf(oacc[mt][3][g] * inv);
        }
    }
}

// ---- kernel 4: out projection + bias + residual (1 M-tile x 4 N-tiles) ----
__global__ void eta_proj_kernel(const bf16* __restrict__ W,     // [512][512]
                                const bf16* __restrict__ outt,  // [b][n][C]
                                const float* __restrict__ bias,
                                const float* __restrict__ x,
                                float* __restrict__ y) {
    int lane = threadIdx.x;
    int half = lane >> 4, lc = lane & 15;
    int nb = blockIdx.x * 64;                    // 4 N tiles of 16
    int mb = blockIdx.y * 16;
    int b  = blockIdx.z;
    v8f acc[4] = {};
    const bf16* arow = W    + (size_t)(mb + lc) * Cc;
    const bf16* bcol = outt + ((size_t)b * Nn + nb + lc) * Cc;
#pragma unroll 2
    for (int cb = 0; cb < Cc; cb += 32) {
        ABu a;
        a.h[0] = *(const v8bf*)(arow + cb + 16 * half);
        a.h[1] = *(const v8bf*)(arow + cb + 8 + 16 * half);
#pragma unroll
        for (int nt = 0; nt < 4; ++nt) {
            v16bf bm = *(const v16bf*)(bcol + (size_t)nt * 16 * Cc + cb + 16 * half);
            acc[nt] = wmma_bf16(a.v, bm, acc[nt]);
        }
    }
#pragma unroll
    for (int nt = 0; nt < 4; ++nt) {
        int t = nb + nt * 16 + lc;
#pragma unroll
        for (int g = 0; g < 8; ++g) {
            int o = mb + g + 8 * half;
            size_t idx = ((size_t)b * Cc + o) * Nn + t;
            y[idx] = acc[nt][g] + bias[o] + x[idx];
        }
    }
}

extern "C" void kernel_launch(void* const* d_in, const int* in_sizes, int n_in,
                              void* d_out, int out_size, void* d_ws, size_t ws_size,
                              hipStream_t stream) {
    const float* x     = (const float*)d_in[0];
    const float* gamma = (const float*)d_in[1];
    const float* beta  = (const float*)d_in[2];
    const float* wqkv  = (const float*)d_in[3];
    const float* wout  = (const float*)d_in[4];
    const float* bout  = (const float*)d_in[5];
    float* y = (float*)d_out;

    char* ws = (char*)d_ws;
    const size_t n_wqkv = (size_t)3 * Cc * Cc;       // 786432
    const size_t n_wout = (size_t)Cc * Cc;           // 262144
    const size_t n_act  = (size_t)Bb * Nn * Cc;      // 4194304 (== b*h*n*64)
    bf16* wqkv_bf = (bf16*)(ws);
    bf16* wout_bf = (bf16*)(ws + 2 * n_wqkv);
    bf16* xnt     = (bf16*)(ws + 2 * (n_wqkv + n_wout));
    bf16* Qb      = (bf16*)(ws + 2 * (n_wqkv + n_wout + n_act));
    bf16* Kb      = (bf16*)(ws + 2 * (n_wqkv + n_wout + 2 * n_act));
    bf16* Vt      = (bf16*)(ws + 2 * (n_wqkv + n_wout + 3 * n_act));
    bf16* outt    = (bf16*)(ws + 2 * (n_wqkv + n_wout + 4 * n_act));

    eta_cvt_kernel<<<(int)((n_wqkv + 255) / 256), 256, 0, stream>>>(
        wqkv, wqkv_bf, (int)n_wqkv, wout, wout_bf, (int)n_wout);

    eta_gn_kernel<<<Bb * Gg, 256, 0, stream>>>(x, gamma, beta, xnt);

    eta_qkv_kernel<<<dim3(Nn / 64, 3 * Cc / 16, Bb), 32, 0, stream>>>(
        wqkv_bf, xnt, Qb, Kb, Vt);

    eta_attn_kernel<<<dim3(Nn / 32, Bb * Hh), 32, 0, stream>>>(Qb, Kb, Vt, outt);

    eta_proj_kernel<<<dim3(Nn / 64, Cc / 16, Bb), 32, 0, stream>>>(
        wout_bf, outt, bout, x, y);
}
